// TransformerXLRelativePositionEmbedding_16441134809291
// MI455X (gfx1250) — matrix operand order repaired and checked by reading, not compiled, not run on hardware
//
#include <hip/hip_runtime.h>

// ---------------------------------------------------------------------------
// TransformerXL relative-position attention logits for MI455X (gfx1250).
// out[b,n,u,w,c] = sum_h Q[b,u,w,n,h]*K[b,u,c,n,h]
//               + (0 <= c-w < 128 ? sum_h Q[b,u,w,n,h]*SIN[c-w,n,h] : 0)
// Strategy: bf16 hi/lo split (bf16x3) on V_WMMA_F32_16X16X32_BF16 so the
// kernel stays HBM-bound (~234 MB traffic @ 23.3 TB/s) instead of being
// limited by the slow f32 16x16x4 WMMA path.
// ---------------------------------------------------------------------------

typedef __attribute__((ext_vector_type(16))) __bf16       v16bf;
typedef __attribute__((ext_vector_type(8)))  float        v8f;
typedef __attribute__((ext_vector_type(4)))  float        v4f;
typedef __attribute__((ext_vector_type(4)))  unsigned int v4u;
typedef __attribute__((ext_vector_type(2)))  unsigned int v2u;

union Frag { v4u q[2]; v16bf v; };

#define DIM_B 4
#define DIM_U 32
#define DIM_W 128
#define DIM_N 8
#define DIM_H 64
#define DIM_C 255
#define DIM_F 128
#define DIM_D 512

__device__ __forceinline__ unsigned short f2bf(float x) {
  unsigned u = __float_as_uint(x);
  u += 0x7FFFu + ((u >> 16) & 1u);          // round-to-nearest-even
  return (unsigned short)(u >> 16);
}
__device__ __forceinline__ float bf2f(unsigned short h) {
  return __uint_as_float(((unsigned)h) << 16);
}

// Split 4 floats into bf16 hi and residual-lo planes, 8-byte packed stores.
__device__ __forceinline__ void split_store(unsigned short* hi, unsigned short* lo,
                                            int off, v4f v) {
  unsigned short h0 = f2bf(v[0]), h1 = f2bf(v[1]), h2 = f2bf(v[2]), h3 = f2bf(v[3]);
  unsigned short l0 = f2bf(v[0] - bf2f(h0));
  unsigned short l1 = f2bf(v[1] - bf2f(h1));
  unsigned short l2 = f2bf(v[2] - bf2f(h2));
  unsigned short l3 = f2bf(v[3] - bf2f(h3));
  v2u ph = { (unsigned)h0 | ((unsigned)h1 << 16), (unsigned)h2 | ((unsigned)h3 << 16) };
  v2u pl = { (unsigned)l0 | ((unsigned)l1 << 16), (unsigned)l2 | ((unsigned)l3 << 16) };
  *(v2u*)(hi + off) = ph;
  *(v2u*)(lo + off) = pl;
}

// ---------------------------------------------------------------------------
// Kernel 1: sinusoidal embedding @ pos_kernel  ->  bf16 hi/lo planes in ws.
// sin_emb[f,n,h] = sum_d sig(127-f, d) * pk[d, n, h];  ws layout [n][f][h].
// Tiny (67 MFLOP) — plain VALU fp32.
// ---------------------------------------------------------------------------
__global__ void __launch_bounds__(256)
sinemb_project_kernel(const float* __restrict__ pk,
                      unsigned short* __restrict__ sin_hi,
                      unsigned short* __restrict__ sin_lo) {
  __shared__ float sig[DIM_D];
  const int f = blockIdx.x;                    // 0..127
  const float p = (float)(127 - f);            // pos = arange(127..0)
  const float kLog = 9.210340371976184f / 255.0f;   // ln(10000)/(num_ts-1)
  for (int d = threadIdx.x; d < DIM_D; d += 256) {
    int i = d & 255;
    float t = p * expf(-(float)i * kLog);
    sig[d] = (d < 256) ? sinf(t) : cosf(t);
  }
  __syncthreads();
  for (int o = threadIdx.x; o < DIM_D; o += 256) {   // o = n*64 + h
    float acc = 0.f;
#pragma unroll 8
    for (int d = 0; d < DIM_D; ++d)
      acc = fmaf(sig[d], pk[(size_t)d * DIM_D + o], acc);
    unsigned short h = f2bf(acc);
    unsigned short l = f2bf(acc - bf2f(h));
    int n = o >> 6, hh = o & 63;
    size_t off = ((size_t)(n * DIM_F + f)) * DIM_H + hh;
    sin_hi[off] = h;
    sin_lo[off] = l;
  }
}

// ---------------------------------------------------------------------------
// Kernel 2: one workgroup per (b,u,n). 256 threads = 8 wave32.
// LDS: Q hi/lo (128x64 bf16), K hi/lo (256x64 bf16, row 255 zero), bd fp32.
// ---------------------------------------------------------------------------
__global__ void __launch_bounds__(256)
relattn_kernel(const float* __restrict__ qg, const float* __restrict__ kg,
               const unsigned short* __restrict__ sin_hi,
               const unsigned short* __restrict__ sin_lo,
               float* __restrict__ out) {
  extern __shared__ char smem[];
  unsigned short* q_hi = (unsigned short*)smem;        // 128*64
  unsigned short* q_lo = q_hi + 128 * 64;
  unsigned short* k_hi = q_lo + 128 * 64;              // 256*64
  unsigned short* k_lo = k_hi + 256 * 64;
  float* bdb = (float*)(k_lo + 256 * 64);              // 128*128 f32

  const int n    = blockIdx.x & 7;
  const int bu   = blockIdx.x >> 3;                    // b*U + u
  const int tid  = threadIdx.x;
  const int lane = tid & 31;
  const int wv   = tid >> 5;                           // 0..7 = M tile
  const int lm   = lane & 15;
  const int lh   = lane >> 4;                          // 0/1 half-wave

  // ---- Phase A: stage Q and K tiles into LDS as bf16 hi/lo planes --------
  const float* qbase = qg + (size_t)bu * (DIM_W * DIM_N * DIM_H) + n * DIM_H;
#pragma unroll
  for (int it = 0; it < 8; ++it) {                     // 128 rows * 16 segs
    int idx = tid + it * 256;
    int row = idx >> 4, seg = (idx & 15) * 4;
    v4f v = *(const v4f*)(qbase + (size_t)row * (DIM_N * DIM_H) + seg);
    split_store(q_hi, q_lo, row * DIM_H + seg, v);
  }
  const float* kbase = kg + (size_t)bu * (DIM_C * DIM_N * DIM_H) + n * DIM_H;
#pragma unroll
  for (int it = 0; it < 16; ++it) {                    // 256 rows * 16 segs
    int idx = tid + it * 256;
    int row = idx >> 4, seg = (idx & 15) * 4;
    v4f v = {};
    if (row < DIM_C) v = *(const v4f*)(kbase + (size_t)row * (DIM_N * DIM_H) + seg);
    split_store(k_hi, k_lo, row * DIM_H + seg, v);
  }
  __syncthreads();

  const int mtile = wv;
  const int arow  = mtile * 16 + lm;                   // A-frag row

  // ---- Phase B: bd[16x128] = Q_tile * sin_emb^T (bf16x3), park in LDS ----
  {
    v8f acc[8] = {};
#pragma unroll
    for (int ks = 0; ks < 2; ++ks) {
      Frag a_hi, a_lo;
      int abk = ks * 32 + lh * 8;                      // A layout per ISA table
      a_hi.q[0] = *(const v4u*)(q_hi + arow * DIM_H + abk);
      a_hi.q[1] = *(const v4u*)(q_hi + arow * DIM_H + abk + 16);
      a_lo.q[0] = *(const v4u*)(q_lo + arow * DIM_H + abk);
      a_lo.q[1] = *(const v4u*)(q_lo + arow * DIM_H + abk + 16);
      int kb = ks * 32 + lh * 8;                       // B layout per ISA table
#pragma unroll
      for (int ft = 0; ft < 8; ++ft) {
        int col = ft * 16 + lm;
        size_t soff = ((size_t)(n * DIM_F + col)) * DIM_H + kb;
        Frag b_hi, b_lo;
        b_hi.q[0] = *(const v4u*)(sin_hi + soff);
        b_hi.q[1] = *(const v4u*)(sin_hi + soff + 16);
        b_lo.q[0] = *(const v4u*)(sin_lo + soff);
        b_lo.q[1] = *(const v4u*)(sin_lo + soff + 16);
        acc[ft] = __builtin_amdgcn_wmma_f32_16x16x32_bf16(false, a_hi.v, false, b_hi.v,
                                                          (short)0, acc[ft], false, false);
        acc[ft] = __builtin_amdgcn_wmma_f32_16x16x32_bf16(false, a_hi.v, false, b_lo.v,
                                                          (short)0, acc[ft], false, false);
        acc[ft] = __builtin_amdgcn_wmma_f32_16x16x32_bf16(false, a_lo.v, false, b_hi.v,
                                                          (short)0, acc[ft], false, false);
      }
    }
#pragma unroll
    for (int ft = 0; ft < 8; ++ft)
#pragma unroll
      for (int j = 0; j < 8; ++j) {                    // C/D layout: M = j + 8*lh
        int w = mtile * 16 + j + lh * 8;
        bdb[w * DIM_F + ft * 16 + lm] = acc[ft][j];
      }
  }
  // No barrier: each wave reads back only its own bd rows.

  // ---- Phase C: ac tiles + fused relative shift + store ------------------
  const int b = bu / DIM_U, u = bu % DIM_U;
  float* outb = out + (size_t)(((b * DIM_N + n) * DIM_U + u)) * DIM_W * DIM_C;

#pragma unroll 1
  for (int half = 0; half < 2; ++half) {               // 2 x 8 C-tiles
    v8f acc[8] = {};
#pragma unroll
    for (int ks = 0; ks < 2; ++ks) {
      Frag a_hi, a_lo;
      int abk = ks * 32 + lh * 8;
      a_hi.q[0] = *(const v4u*)(q_hi + arow * DIM_H + abk);
      a_hi.q[1] = *(const v4u*)(q_hi + arow * DIM_H + abk + 16);
      a_lo.q[0] = *(const v4u*)(q_lo + arow * DIM_H + abk);
      a_lo.q[1] = *(const v4u*)(q_lo + arow * DIM_H + abk + 16);
      int kb = ks * 32 + lh * 8;
#pragma unroll
      for (int ci = 0; ci < 8; ++ci) {
        int col = (half * 8 + ci) * 16 + lm;
        Frag b_hi, b_lo;
        b_hi.q[0] = *(const v4u*)(k_hi + col * DIM_H + kb);
        b_hi.q[1] = *(const v4u*)(k_hi + col * DIM_H + kb + 16);
        b_lo.q[0] = *(const v4u*)(k_lo + col * DIM_H + kb);
        b_lo.q[1] = *(const v4u*)(k_lo + col * DIM_H + kb + 16);
        acc[ci] = __builtin_amdgcn_wmma_f32_16x16x32_bf16(false, a_hi.v, false, b_hi.v,
                                                          (short)0, acc[ci], false, false);
        acc[ci] = __builtin_amdgcn_wmma_f32_16x16x32_bf16(false, a_hi.v, false, b_lo.v,
                                                          (short)0, acc[ci], false, false);
        acc[ci] = __builtin_amdgcn_wmma_f32_16x16x32_bf16(false, a_lo.v, false, b_hi.v,
                                                          (short)0, acc[ci], false, false);
      }
    }
#pragma unroll
    for (int ci = 0; ci < 8; ++ci) {
      int c = (half * 8 + ci) * 16 + lm;
#pragma unroll
      for (int j = 0; j < 8; ++j) {
        int w = mtile * 16 + j + lh * 8;
        float v = acc[ci][j];
        int fidx = c - w;                              // rel-shift gather
        if ((unsigned)fidx < 128u) v += bdb[w * DIM_F + fidx];
        if (c < DIM_C) outb[(size_t)w * DIM_C + c] = v;
      }
    }
  }
}

// ---------------------------------------------------------------------------
extern "C" void kernel_launch(void* const* d_in, const int* in_sizes, int n_in,
                              void* d_out, int out_size, void* d_ws, size_t ws_size,
                              hipStream_t stream) {
  (void)in_sizes; (void)n_in; (void)out_size; (void)ws_size;
  const float* q  = (const float*)d_in[0];
  const float* k  = (const float*)d_in[1];
  const float* pk = (const float*)d_in[2];
  float* out = (float*)d_out;

  unsigned short* sin_hi = (unsigned short*)d_ws;                 // 8*128*64 u16
  unsigned short* sin_lo = sin_hi + DIM_N * DIM_F * DIM_H;

  const int lds_bytes = (128 * 64 + 128 * 64 + 256 * 64 + 256 * 64) * 2 + 128 * 128 * 4; // 160 KB
  (void)hipFuncSetAttribute((const void*)relattn_kernel,
                            hipFuncAttributeMaxDynamicSharedMemorySize, lds_bytes);

  sinemb_project_kernel<<<dim3(DIM_F), dim3(256), 0, stream>>>(pk, sin_hi, sin_lo);
  relattn_kernel<<<dim3(DIM_B * DIM_U * DIM_N), dim3(256), lds_bytes, stream>>>(
      q, k, sin_hi, sin_lo, out);
}
